// LRCP_24077586661553
// MI455X (gfx1250) — matrix operand order repaired and checked by reading, not compiled, not run on hardware
//
#include <hip/hip_runtime.h>
#include <stdint.h>
#include <stddef.h>

#define DCH   256
#define BATCH 128
#define NPIX  784
#define KPAD  800        // 784 padded to K-step multiple with zeros
#define FEATN 32896      // 256*257/2 ; == 1028 * 32
#define NCLS  200

#define BM 128
#define BN 128
#define BK 32
#define LDT 20           // dwords per LDS tile row (16 used + 4 pad; 80B stride)
#define NNSTEPS  (DCH / BK)              // 8
#define COVSTEPS (KPAD / BK)             // 25
#define FCCHUNK  4
#define FCSTEPS  (FEATN / BK / FCCHUNK)  // 257

typedef __attribute__((ext_vector_type(16))) __bf16 v16bf;
typedef __attribute__((ext_vector_type(8)))  float  v8f;
typedef __attribute__((ext_vector_type(4)))  uint32_t tdm_v4u;
typedef __attribute__((ext_vector_type(8)))  int      tdm_v8i;
typedef __attribute__((ext_vector_type(4)))  int      tdm_v4i;

union Frag { v16bf v; uint32_t w[8]; };
union Acc  { v8f v; float f[8]; };
union Q16  { uint4 q; unsigned short h[8]; };

__device__ __forceinline__ unsigned short f2bf(float x) {
  union { float f; uint32_t u; } c; c.f = x;
  uint32_t u = c.u;
  return (unsigned short)((u + 0x7FFFu + ((u >> 16) & 1u)) >> 16);  // RNE
}

// ---------------------------------------------------------------------------
// Tensor Data Mover: one-instruction 2D bf16 tile load Global->LDS.
// LDS pad: after every 16 dwords (one 64B tile row) insert 4 dwords
// -> row stride 80B == LDT*4, identical to the manual staging layout.
// ---------------------------------------------------------------------------
#if defined(__has_builtin)
#  if __has_builtin(__builtin_amdgcn_s_wait_tensorcnt)
#    define WAIT_TENSOR() __builtin_amdgcn_s_wait_tensorcnt(0)
#  endif
#endif
#ifndef WAIT_TENSOR
#  define WAIT_TENSOR() asm volatile("s_wait_tensorcnt 0x0" ::: "memory")
#endif

__device__ __forceinline__ void tdm_tile_2d(uint32_t lds_byte, const void* gptr,
                                            uint32_t tile_w, uint32_t tile_h,
                                            uint32_t tens_w, uint32_t tens_h,
                                            uint32_t stride_elems) {
  const uint64_t ga = (uint64_t)(uintptr_t)gptr;
  // D# group0: count=1 | lds_addr | global_addr[56:0] | type=2
  tdm_v4u g0 = {1u, lds_byte, (uint32_t)ga,
                (uint32_t)((ga >> 32) & 0x1FFFFFFu) | (2u << 30)};
  // D# group1: data_size=1(2B), pad_enable, pad_interval=3(16dw), pad_amount=3(4dw)
  tdm_v8i g1 = {(int)((1u << 16) | (1u << 20) | (3u << 22) | (3u << 25)),
                (int)((tens_w & 0xFFFFu) << 16),
                (int)((tens_w >> 16) | ((tens_h & 0xFFFFu) << 16)),
                (int)((tens_h >> 16) | (tile_w << 16)),
                (int)(tile_h & 0xFFFFu),
                (int)stride_elems, 0, 0};
  tdm_v4i z4 = {0, 0, 0, 0};
#if __clang_major__ >= 23
  tdm_v8i z8 = {0, 0, 0, 0, 0, 0, 0, 0};
  __builtin_amdgcn_tensor_load_to_lds(g0, g1, z4, z4, z8, 0);
#else
  __builtin_amdgcn_tensor_load_to_lds(g0, g1, z4, z4, 0);
#endif
}

// 64x32 wave tile: 4 A-frags x 2 B-frags = 8 WMMAs per K-step
__device__ __forceinline__ void mma_step(const uint32_t* __restrict__ As,
                                         const uint32_t* __restrict__ Bt,
                                         int wm, int wn, int m16, int half,
                                         Acc (&acc)[4][2]) {
  Frag fa[4], fb[2];
#pragma unroll
  for (int fm = 0; fm < 4; ++fm) {
    const uint32_t* src = &As[(wm + fm * 16 + m16) * LDT];
#pragma unroll
    for (int v = 0; v < 8; ++v)
      fa[fm].w[v] = src[((v >> 2) * 8) + (v & 3) + half * 4];
  }
#pragma unroll
  for (int fn = 0; fn < 2; ++fn) {
    const uint32_t* src = &Bt[(wn + fn * 16 + m16) * LDT];
#pragma unroll
    for (int v = 0; v < 8; ++v) fb[fn].w[v] = src[((v >> 2) * 8) + (v & 3) + half * 4];
  }
#pragma unroll
  for (int fm = 0; fm < 4; ++fm)
#pragma unroll
    for (int fn = 0; fn < 2; ++fn)
      acc[fm][fn].v = __builtin_amdgcn_wmma_f32_16x16x32_bf16(
          false, fa[fm].v, false, fb[fn].v, (short)0, acc[fm][fn].v, false, false);
}

// ---------------------------------------------------------------------------
// attention head + fused f32->bf16 conversion of pro (padded 784->800)
// ---------------------------------------------------------------------------
__global__ __launch_bounds__(256) void k_att(const float* __restrict__ pro,
                                             const float* __restrict__ w1,
                                             const float* __restrict__ w2,
                                             float* __restrict__ gbuf,
                                             float* __restrict__ satt,
                                             unsigned short* __restrict__ Xbf) {
  __shared__ float gs[DCH];
  __shared__ float hs[16];
  __shared__ float as_[DCH];
  const int b = blockIdx.x, t = threadIdx.x;
  const int wave = t >> 5, lane = t & 31;
  const float* pb = pro + (size_t)b * DCH * NPIX;
  unsigned short* xb = Xbf + (size_t)b * DCH * KPAD;
  for (int ci = 0; ci < 32; ++ci) {
    const int ch = wave * 32 + ci;
    const float* row = pb + (size_t)ch * NPIX;
    unsigned short* xr = xb + (size_t)ch * KPAD;
    float s = 0.0f;
    for (int p = lane; p < KPAD; p += 32) {
      const float v = (p < NPIX) ? row[p] : 0.0f;
      s += v;
      xr[p] = f2bf(v);
    }
    for (int off = 16; off > 0; off >>= 1) s += __shfl_xor(s, off, 32);
    if (lane == 0) gs[ch] = s * (1.0f / (float)NPIX);
  }
  __syncthreads();
  gbuf[(size_t)b * DCH + t] = gs[t];
  if (t < 16) {
    float s = 0.0f;
    for (int c = 0; c < DCH; ++c) s += w1[t * DCH + c] * gs[c];
    hs[t] = s;
  }
  __syncthreads();
  {
    float s = 0.0f;
    for (int o = 0; o < 16; ++o) s += w2[t * 16 + o] * hs[o];
    as_[t] = 1.0f / (1.0f + expf(-s));
  }
  __syncthreads();
  const float av = as_[t];
  float* ob = satt + (size_t)b * DCH * DCH;
  for (int j = 0; j < DCH; ++j) ob[(size_t)j * DCH + t] = as_[j] * av;
}

// ---------------------------------------------------------------------------
__global__ __launch_bounds__(256) void k_cvt_w(const float* __restrict__ fcw,
                                               unsigned short* __restrict__ fcwbf) {
  const size_t total = (size_t)NCLS * FEATN;
  for (size_t i = (size_t)blockIdx.x * blockDim.x + threadIdx.x; i < total;
       i += (size_t)gridDim.x * blockDim.x)
    fcwbf[i] = f2bf(fcw[i]);
}

// ---------------------------------------------------------------------------
// Batched NN GEMM 256^3: A-tile via TDM, B-tile transpose-on-store.
//  MODE 0: bf16 out (Z = ZY @ Z)
//  MODE 1: ZY = 0.5*(3I - acc), bf16 out (T = Z @ Y)
//  MODE 2: acc = Ynew; fused L-update + next-iteration J/Y update -> Ybf out
//  MODE 3: final: feat[tril] = acc * sqrt(normA), bf16 out
// ---------------------------------------------------------------------------
template <int MODE>
__global__ __launch_bounds__(256) void k_gemm_nn(
    const unsigned short* __restrict__ Amat,
    const unsigned short* __restrict__ Bmat,
    unsigned short* __restrict__ outB,
    float* __restrict__ J1, float* __restrict__ J2,
    float* __restrict__ L1, float* __restrict__ L2,
    const float* __restrict__ satt, const float* __restrict__ normA,
    float aux, float mu1c, float mu2c,
    float eta1n, float b1e1n, float e1m1n,
    float eta2n, float b2e2n, float e2m2n,
    float m1e3n, float m2e3n, float eta3n) {
  __shared__ uint32_t As[2][BM * LDT];
  __shared__ uint32_t Bt[2][BN * LDT];
  const int b  = blockIdx.z;
  const int tm = blockIdx.y * BM;
  const int tn = blockIdx.x * BN;
  const unsigned short* Ab = Amat + (size_t)b * DCH * DCH;
  const unsigned short* Bb = Bmat + (size_t)b * DCH * DCH;
  const int t = threadIdx.x, lane = t & 31, wave = t >> 5;
  const int wm = (wave >> 2) * 64, wn = (wave & 3) * 32;
  const int m16 = lane & 15, half = lane >> 4;

  Acc acc[4][2];
#pragma unroll
  for (int i = 0; i < 4; ++i)
#pragma unroll
    for (int j = 0; j < 2; ++j)
#pragma unroll
      for (int q = 0; q < 8; ++q) acc[i][j].f[q] = 0.0f;

  const int bk = t >> 3, bn0 = (t & 7) * 16;  // B: transpose-on-store
  Q16 rb0, rb1;
  auto loadgB = [&](int kk) {
    const uint4* gb = (const uint4*)(Bb + (size_t)(kk + bk) * DCH + tn + bn0);
    rb0.q = gb[0];
    rb1.q = gb[1];
  };
  auto storesB = [&](int buf) {
    unsigned short* bt16 = (unsigned short*)Bt[buf];
#pragma unroll
    for (int i = 0; i < 8; ++i) bt16[(bn0 + i) * (LDT * 2) + bk] = rb0.h[i];
#pragma unroll
    for (int i = 0; i < 8; ++i) bt16[(bn0 + 8 + i) * (LDT * 2) + bk] = rb1.h[i];
  };
  const uint32_t ldsA[2] = {(uint32_t)(uintptr_t)&As[0][0],
                            (uint32_t)(uintptr_t)&As[1][0]};
  const unsigned short* Atile = Ab + (size_t)tm * DCH;

  if (wave == 0) tdm_tile_2d(ldsA[0], Atile, BK, BM, BK, BM, DCH);
  loadgB(0);
  storesB(0);
  if (wave == 0) WAIT_TENSOR();
  __syncthreads();
  for (int s = 0; s < NNSTEPS; ++s) {
    const int cur = s & 1;
    const bool nxt = (s + 1) < NNSTEPS;
    if (nxt) {
      if (wave == 0)
        tdm_tile_2d(ldsA[cur ^ 1], Atile + (s + 1) * BK, BK, BM, BK, BM, DCH);
      loadgB((s + 1) * BK);
    }
    mma_step(As[cur], Bt[cur], wm, wn, m16, half, acc);
    if (nxt) {
      storesB(cur ^ 1);
      if (wave == 0) WAIT_TENSOR();
    }
    __syncthreads();
  }

  const size_t ob = (size_t)b * DCH * DCH;
  float sA = 0.0f;
  if (MODE == 3) sA = sqrtf(normA[b]);
#pragma unroll
  for (int fm = 0; fm < 4; ++fm)
#pragma unroll
    for (int fn = 0; fn < 2; ++fn)
#pragma unroll
      for (int r = 0; r < 8; ++r) {
        const int row = tm + wm + fm * 16 + half * 8 + r;
        const int col = tn + wn + fn * 16 + m16;
        float v = acc[fm][fn].f[r];
        const size_t o = ob + (size_t)row * DCH + col;
        if (MODE == 0) {
          outB[o] = f2bf(v);
        } else if (MODE == 1) {
          v = 0.5f * ((row == col ? 3.0f : 0.0f) - v);
          outB[o] = f2bf(v);
        } else if (MODE == 2) {
          float j1 = J1[o], j2 = J2[o], l1 = L1[o], l2 = L2[o];
          l1 = aux * l1 + mu1c * (j1 - v);              // L update (uses Ynew)
          l2 = aux * l2 + mu2c * (j2 - v);
          j1 = j1 - e1m1n * (j1 - v) - eta1n * l1;      // next-iter J update
          if (row == col) j1 *= (1.0f - b1e1n);
          j2 = j2 - e2m2n * (j2 - v) - eta2n * l2;
          j2 *= (1.0f - b2e2n * (1.0f - satt[o]));
          const float yn = v + m1e3n * (j1 - v) + m2e3n * (j2 - v) + eta3n * (l1 + l2);
          L1[o] = l1; L2[o] = l2; J1[o] = j1; J2[o] = j2;
          outB[o] = f2bf(yn);
        } else {  // MODE 3: tril gather + sqrt(trace) scaling, bf16 features
          if (col <= row)
            outB[(size_t)b * FEATN + (row * (row + 1)) / 2 + col] = f2bf(v * sA);
        }
      }
}

// ---------------------------------------------------------------------------
// Covariance (NT, K=800 zero-padded): A = X Xᵀ/784 - g gᵀ ; both tiles via TDM
// ---------------------------------------------------------------------------
__global__ __launch_bounds__(256) void k_gemm_cov(const unsigned short* __restrict__ X,
                                                  const float* __restrict__ gbuf,
                                                  float* __restrict__ outA) {
  __shared__ uint32_t As[2][BM * LDT];
  __shared__ uint32_t Bt[2][BN * LDT];
  const int b  = blockIdx.z;
  const int tm = blockIdx.y * BM;
  const int tn = blockIdx.x * BN;
  const unsigned short* Xb = X + (size_t)b * DCH * KPAD;
  const int t = threadIdx.x, lane = t & 31, wave = t >> 5;
  const int wm = (wave >> 2) * 64, wn = (wave & 3) * 32;
  const int m16 = lane & 15, half = lane >> 4;

  Acc acc[4][2];
#pragma unroll
  for (int i = 0; i < 4; ++i)
#pragma unroll
    for (int j = 0; j < 2; ++j)
#pragma unroll
      for (int q = 0; q < 8; ++q) acc[i][j].f[q] = 0.0f;

  const uint32_t ldsA[2] = {(uint32_t)(uintptr_t)&As[0][0],
                            (uint32_t)(uintptr_t)&As[1][0]};
  const uint32_t ldsB[2] = {(uint32_t)(uintptr_t)&Bt[0][0],
                            (uint32_t)(uintptr_t)&Bt[1][0]};
  const unsigned short* Atile = Xb + (size_t)tm * KPAD;
  const unsigned short* Btile = Xb + (size_t)tn * KPAD;

  if (wave == 0) {
    tdm_tile_2d(ldsA[0], Atile, BK, BM, BK, BM, KPAD);
    tdm_tile_2d(ldsB[0], Btile, BK, BM, BK, BM, KPAD);
    WAIT_TENSOR();
  }
  __syncthreads();
  for (int s = 0; s < COVSTEPS; ++s) {
    const int cur = s & 1;
    const bool nxt = (s + 1) < COVSTEPS;
    if (nxt && wave == 0) {
      tdm_tile_2d(ldsA[cur ^ 1], Atile + (s + 1) * BK, BK, BM, BK, BM, KPAD);
      tdm_tile_2d(ldsB[cur ^ 1], Btile + (s + 1) * BK, BK, BM, BK, BM, KPAD);
    }
    mma_step(As[cur], Bt[cur], wm, wn, m16, half, acc);
    if (nxt && wave == 0) WAIT_TENSOR();
    __syncthreads();
  }

  const size_t ob = (size_t)b * DCH * DCH;
  const float* gv = gbuf + (size_t)b * DCH;
#pragma unroll
  for (int fm = 0; fm < 4; ++fm)
#pragma unroll
    for (int fn = 0; fn < 2; ++fn)
#pragma unroll
      for (int r = 0; r < 8; ++r) {
        const int row = tm + wm + fm * 16 + half * 8 + r;
        const int col = tn + wn + fn * 16 + m16;
        outA[ob + (size_t)row * DCH + col] =
            acc[fm][fn].f[r] * (1.0f / (float)NPIX) - gv[row] * gv[col];
      }
}

// ---------------------------------------------------------------------------
// FC split-K GEMM: part[chunk] = feat[128x32896] . fcwᵀ ; tiles via TDM,
// class rows >= 200 zero-filled by TDM OOB semantics (tensor_dim1 = NCLS - tn)
// ---------------------------------------------------------------------------
__global__ __launch_bounds__(256) void k_gemm_fc(const unsigned short* __restrict__ featbf,
                                                 const unsigned short* __restrict__ fcwbf,
                                                 float* __restrict__ part) {
  __shared__ uint32_t As[2][BM * LDT];
  __shared__ uint32_t Bt[2][BN * LDT];
  const int chunk = blockIdx.z;
  const int tn = blockIdx.x * BN;  // 0 or 128 (cols 200..255 dead)
  const int t = threadIdx.x, lane = t & 31, wave = t >> 5;
  const int wm = (wave >> 2) * 64, wn = (wave & 3) * 32;
  const int m16 = lane & 15, half = lane >> 4;

  Acc acc[4][2];
#pragma unroll
  for (int i = 0; i < 4; ++i)
#pragma unroll
    for (int j = 0; j < 2; ++j)
#pragma unroll
      for (int q = 0; q < 8; ++q) acc[i][j].f[q] = 0.0f;

  const uint32_t ldsA[2] = {(uint32_t)(uintptr_t)&As[0][0],
                            (uint32_t)(uintptr_t)&As[1][0]};
  const uint32_t ldsB[2] = {(uint32_t)(uintptr_t)&Bt[0][0],
                            (uint32_t)(uintptr_t)&Bt[1][0]};
  const uint32_t bvalid = (uint32_t)(NCLS - tn);  // rows past this are OOB-zero
  const int k0 = chunk * FCSTEPS * BK;
  const unsigned short* Atile = featbf + k0;
  const unsigned short* Btile = fcwbf + (size_t)tn * FEATN + k0;

  if (wave == 0) {
    tdm_tile_2d(ldsA[0], Atile, BK, BM, BK, BM, FEATN);
    tdm_tile_2d(ldsB[0], Btile, BK, BM, BK, bvalid, FEATN);
    WAIT_TENSOR();
  }
  __syncthreads();
  for (int s = 0; s < FCSTEPS; ++s) {
    const int cur = s & 1;
    const bool nxt = (s + 1) < FCSTEPS;
    if (nxt && wave == 0) {
      tdm_tile_2d(ldsA[cur ^ 1], Atile + (s + 1) * BK, BK, BM, BK, BM, FEATN);
      tdm_tile_2d(ldsB[cur ^ 1], Btile + (s + 1) * BK, BK, BM, BK, bvalid, FEATN);
    }
    mma_step(As[cur], Bt[cur], wm, wn, m16, half, acc);
    if (nxt && wave == 0) WAIT_TENSOR();
    __syncthreads();
  }

#pragma unroll
  for (int fm = 0; fm < 4; ++fm)
#pragma unroll
    for (int fn = 0; fn < 2; ++fn)
#pragma unroll
      for (int r = 0; r < 8; ++r) {
        const int row = wm + fm * 16 + half * 8 + r;                 // batch
        const int col = tn + wn + fn * 16 + m16;                     // class
        part[((size_t)chunk * BM + row) * 256 + col] = acc[fm][fn].f[r];
      }
}

__global__ __launch_bounds__(256) void k_fcred(const float* __restrict__ part,
                                               const float* __restrict__ fcb,
                                               float* __restrict__ out) {
  const int i = blockIdx.x * 256 + threadIdx.x;
  if (i >= BATCH * NCLS) return;
  const int b = i / NCLS, o = i - b * NCLS;
  float s = fcb[o];
#pragma unroll
  for (int c = 0; c < FCCHUNK; ++c) s += part[((size_t)c * BATCH + b) * 256 + o];
  out[i] = s;
}

// ---------------------------------------------------------------------------
__global__ __launch_bounds__(256) void k_trace(const float* __restrict__ A,
                                               float* __restrict__ normA) {
  __shared__ float s[256];
  const int b = blockIdx.x, t = threadIdx.x;
  s[t] = A[(size_t)b * DCH * DCH + (size_t)t * (DCH + 1)];
  __syncthreads();
  for (int off = 128; off > 0; off >>= 1) {
    if (t < off) s[t] += s[t + off];
    __syncthreads();
  }
  if (t == 0) normA[b] = s[0];
}

// iteration-0 elementwise construction
__global__ __launch_bounds__(256) void k_init(const float* __restrict__ A,
                                              const float* __restrict__ satt,
                                              const float* __restrict__ normA,
                                              float* __restrict__ J1, float* __restrict__ J2,
                                              unsigned short* __restrict__ Ybf,
                                              unsigned short* __restrict__ ZYbf,
                                              unsigned short* __restrict__ Zbf0,
                                              float e1b1, float e2b2, float m1e3, float m2e3) {
  const size_t total = (size_t)BATCH * DCH * DCH;
  for (size_t i = (size_t)blockIdx.x * blockDim.x + threadIdx.x; i < total;
       i += (size_t)gridDim.x * blockDim.x) {
    const int b = (int)(i >> 16);
    const int r = (int)(i >> 8) & 255, c = (int)i & 255;
    const float a = A[i] / normA[b];
    const bool diag = (r == c);
    const float j1 = a - e1b1 * (diag ? a : 0.0f);
    const float j2 = a - e2b2 * (1.0f - satt[i]) * a;
    const float y = a + m1e3 * (j1 - a) + m2e3 * (j2 - a);
    const float zy = 0.5f * ((diag ? 3.0f : 0.0f) - a);
    J1[i] = j1;
    J2[i] = j2;
    Ybf[i] = f2bf(y);
    const unsigned short z = f2bf(zy);
    ZYbf[i] = z;
    Zbf0[i] = z;
  }
}

// ---------------------------------------------------------------------------
extern "C" void kernel_launch(void* const* d_in, const int* in_sizes, int n_in,
                              void* d_out, int out_size, void* d_ws, size_t ws_size,
                              hipStream_t stream) {
  (void)in_sizes; (void)n_in; (void)out_size; (void)ws_size;
  const float* pro = (const float*)d_in[0];
  const float* w1  = (const float*)d_in[1];
  const float* w2  = (const float*)d_in[2];
  const float* fcw = (const float*)d_in[3];
  const float* fcb = (const float*)d_in[4];

  float* out_cls  = (float*)d_out;
  float* out_A    = out_cls + (size_t)BATCH * NCLS;
  float* out_satt = out_A + (size_t)BATCH * DCH * DCH;

  char* wp = (char*)d_ws;
  auto take = [&](size_t bytes) -> char* {
    char* p = wp;
    wp += (bytes + 255) & ~(size_t)255;
    return p;
  };
  const size_t MATF = (size_t)BATCH * DCH * DCH * sizeof(float);
  const size_t MATB = (size_t)BATCH * DCH * DCH * sizeof(unsigned short);
  unsigned short* Xbf   = (unsigned short*)take((size_t)BATCH * DCH * KPAD * 2);
  float* gbuf           = (float*)take((size_t)BATCH * DCH * 4);
  float* normA          = (float*)take((size_t)BATCH * 4);
  unsigned short* Yb0   = (unsigned short*)take(MATB);
  unsigned short* Yb1   = (unsigned short*)take(MATB);
  unsigned short* ZYbf  = (unsigned short*)take(MATB);
  unsigned short* Zbf0  = (unsigned short*)take(MATB);
  unsigned short* Zbf1  = (unsigned short*)take(MATB);
  float* J1 = (float*)take(MATF);
  float* J2 = (float*)take(MATF);
  float* L1 = (float*)take(MATF);
  float* L2 = (float*)take(MATF);
  unsigned short* featbf = (unsigned short*)take((size_t)BATCH * FEATN * 2);
  unsigned short* fcwbf  = (unsigned short*)take((size_t)NCLS * FEATN * 2);
  float* part            = (float*)take((size_t)FCCHUNK * BATCH * 256 * 4);
  unsigned short* Yb[2] = {Yb0, Yb1};
  unsigned short* Zb[2] = {Zbf0, Zbf1};

  const dim3 eb(256);
  const dim3 gg(DCH / BN, DCH / BM, BATCH);  // 2 x 2 x 128

  k_att<<<BATCH, eb, 0, stream>>>(pro, w1, w2, gbuf, out_satt, Xbf);
  k_cvt_w<<<4096, eb, 0, stream>>>(fcw, fcwbf);
  k_gemm_cov<<<gg, eb, 0, stream>>>(Xbf, gbuf, out_A);
  k_trace<<<BATCH, eb, 0, stream>>>(out_A, normA);

  const double BETA1 = 1e-3, BETA2 = 1e-3, ROPH = 1.05;
  double mu1 = 0.5, mu2 = 0.5;  // mu of the *current* iteration
  {
    const double e1 = 1.0 / mu1, e2 = 1.0 / mu2, e3 = 1.0 / (mu1 + mu2);
    k_init<<<4096, eb, 0, stream>>>(out_A, out_satt, normA, J1, J2, Yb[0], ZYbf, Zb[0],
                                    (float)(e1 * BETA1), (float)(e2 * BETA2),
                                    (float)(mu1 * e3), (float)(mu2 * e3));
  }
  int yc = 0, zc = 0;
  auto launch_Y = [&](float aux) {  // Y-GEMM with fused L-update + next J/Y update
    const double m1n = mu1 * ROPH, m2n = mu2 * ROPH;
    const double e1n = 1.0 / m1n, e2n = 1.0 / m2n, e3n = 1.0 / (m1n + m2n);
    k_gemm_nn<2><<<gg, eb, 0, stream>>>(
        Yb[yc], ZYbf, Yb[1 - yc], J1, J2, L1, L2, out_satt, (const float*)nullptr,
        aux, (float)mu1, (float)mu2,
        (float)e1n, (float)(BETA1 * e1n), (float)(e1n * m1n),
        (float)e2n, (float)(BETA2 * e2n), (float)(e2n * m2n),
        (float)(m1n * e3n), (float)(m2n * e3n), (float)e3n);
    yc ^= 1;
    mu1 = m1n;
    mu2 = m2n;
  };

  // iteration 0: Y1 = Ymid @ ZY0 (+ fused L aux=0 and iter-1 J/Y update)
  launch_Y(0.0f);
  for (int i = 1; i < 7; ++i) {
    // T = Z @ Y  ->  ZY = 0.5*(3I - T)  (bf16)
    k_gemm_nn<1><<<gg, eb, 0, stream>>>(Zb[zc], Yb[yc], ZYbf,
                                        (float*)nullptr, (float*)nullptr,
                                        (float*)nullptr, (float*)nullptr,
                                        (const float*)nullptr, (const float*)nullptr,
                                        0, 0, 0, 0, 0, 0, 0, 0, 0, 0, 0, 0);
    if (i < 6) {
      // Z = ZY @ Z  (bf16)
      k_gemm_nn<0><<<gg, eb, 0, stream>>>(ZYbf, Zb[zc], Zb[1 - zc],
                                          (float*)nullptr, (float*)nullptr,
                                          (float*)nullptr, (float*)nullptr,
                                          (const float*)nullptr, (const float*)nullptr,
                                          0, 0, 0, 0, 0, 0, 0, 0, 0, 0, 0, 0);
      zc ^= 1;
      launch_Y(0.9f);
    } else {
      // final Y-GEMM: write tril features scaled by sqrt(trace), bf16
      k_gemm_nn<3><<<gg, eb, 0, stream>>>(Yb[yc], ZYbf, featbf,
                                          (float*)nullptr, (float*)nullptr,
                                          (float*)nullptr, (float*)nullptr,
                                          (const float*)nullptr, normA,
                                          0, 0, 0, 0, 0, 0, 0, 0, 0, 0, 0, 0);
    }
  }

  const dim3 fg(2, 1, FCCHUNK);
  k_gemm_fc<<<fg, eb, 0, stream>>>(featbf, fcwbf, part);
  k_fcred<<<(BATCH * NCLS + 255) / 256, eb, 0, stream>>>(part, fcb, out_cls);
}
